// CustomGRU_28389733827227
// MI455X (gfx1250) — matrix operand order repaired and checked by reading, not compiled, not run on hardware
//
#include <hip/hip_runtime.h>
#include <hip/hip_bf16.h>
#include <math.h>

typedef __attribute__((ext_vector_type(16))) __bf16 v16bf;
typedef __attribute__((ext_vector_type(8)))  float  v8f;

#define B_  64
#define T_  512
#define D_  512
#define U_  1024
#define BT_ (B_ * T_)
#define SCAN_BLOCKS 32
#define BLOCK_THREADS 256

#define WMMA_BF16(acc, a, b)                                                   \
    (acc) = __builtin_amdgcn_wmma_f32_16x16x32_bf16(false, (a), false, (b),    \
                                                    (short)0, (acc), false, false)

// ---------------------------------------------------------------------------
// WMMA 16-bit A fragment layout (16x32 MxK, wave32), stored per (kt,lane) as
// 16 contiguous bf16 (32 B):
//   lanes 0-15 : M=lane,    halves 0-7 => K=k0+0..7,  halves 8-15 => K=k0+16..23
//   lanes 16-31: M=lane-16, halves 0-7 => K=k0+8..15, halves 8-15 => K=k0+24..31
// ---------------------------------------------------------------------------

// Cooperative block staging: convert a 16-row fp32 strip (ld-major) into the
// bf16 A-fragment image in LDS.  256 threads, each owns (ktiles*32)/256
// fragment entries; one 32-byte LDS store per entry.
static __device__ __forceinline__ void stage_a(__bf16* __restrict__ lds,
                                               const float* __restrict__ A,
                                               int rowBase, int ld, int ktiles) {
    for (int e = threadIdx.x; e < ktiles * 32; e += BLOCK_THREADS) {
        const int kt = e >> 5;
        const int lane = e & 31;
        const float* r = A + (size_t)(rowBase + (lane & 15)) * ld + kt * 32 +
                         ((lane & 16) ? 8 : 0);
        v16bf a;
#pragma unroll
        for (int j = 0; j < 8; ++j) {
            a[j]     = (__bf16)r[j];
            a[j + 8] = (__bf16)r[j + 16];
        }
        *(v16bf*)(lds + (size_t)e * 16) = a;
    }
}

// Same, but the staged operand is the elementwise product r * h_prev.
static __device__ __forceinline__ void stage_a_mul(__bf16* __restrict__ lds,
                                                   const float* __restrict__ A,
                                                   const float* __restrict__ M,
                                                   int rowBase, int ld, int ktiles) {
    for (int e = threadIdx.x; e < ktiles * 32; e += BLOCK_THREADS) {
        const int kt = e >> 5;
        const int lane = e & 31;
        const size_t off = (size_t)(rowBase + (lane & 15)) * ld + kt * 32 +
                           ((lane & 16) ? 8 : 0);
        const float* a0 = A + off;
        const float* m0 = M + off;
        v16bf a;
#pragma unroll
        for (int j = 0; j < 8; ++j) {
            a[j]     = (__bf16)(a0[j] * m0[j]);
            a[j + 8] = (__bf16)(a0[j + 16] * m0[j + 16]);
        }
        *(v16bf*)(lds + (size_t)e * 16) = a;
    }
}

static __device__ __forceinline__ float sigmoidf_(float x) {
    return 1.0f / (1.0f + __expf(-x));
}

// Device-wide split barrier: one representative thread per block arrives at a
// monotonically increasing target.  Counter is zeroed by hipMemsetAsync before
// each launch, so graph replays are deterministic.
static __device__ __forceinline__ void grid_barrier(unsigned* __restrict__ ctr,
                                                    unsigned target) {
    __syncthreads();
    if (threadIdx.x == 0) {
        __threadfence();   // make this block's global writes visible device-wide
        __hip_atomic_fetch_add(ctr, 1u, __ATOMIC_RELEASE, __HIP_MEMORY_SCOPE_AGENT);
        while (__hip_atomic_load(ctr, __ATOMIC_ACQUIRE, __HIP_MEMORY_SCOPE_AGENT) < target) {
            __builtin_amdgcn_s_sleep(2);
        }
    }
    __syncthreads();
}

// ---------------------------------------------------------------------------
// Pack a [Ktot x Ntot] row-major fp32 matrix into bf16 B-fragment tiles for
// v_wmma_f32_16x16x32_bf16.  Tile (kt,nt): 32 lanes x 16 contiguous bf16.
//   lanes 0-15 : N=nt*16+lane,    halves 0-7 => K=kt*32+0..7,  halves 8-15 => K=kt*32+16..23
//   lanes 16-31: N=nt*16+lane-16, halves 0-7 => K=kt*32+8..15, halves 8-15 => K=kt*32+24..31
// ---------------------------------------------------------------------------
__global__ void pack_b_kernel(const float* __restrict__ src, __bf16* __restrict__ dst,
                              int Ktot, int Ntot) {
    const int ktiles = Ktot >> 5;
    const int ntiles = Ntot >> 4;
    const int tid = blockIdx.x * blockDim.x + threadIdx.x;
    const int total = ktiles * ntiles * 32;
    if (tid >= total) return;
    const int lane = tid & 31;
    const int tile = tid >> 5;
    const int nt = tile % ntiles;
    const int kt = tile / ntiles;
    const int n = nt * 16 + (lane & 15);
    const int kbase = kt * 32 + ((lane & 16) ? 8 : 0);
    __bf16* d = dst + (size_t)tid * 16;
#pragma unroll
    for (int j = 0; j < 16; ++j)
        d[j] = (__bf16)src[(size_t)(kbase + (j < 8 ? j : j + 8)) * Ntot + n];
}

// ---------------------------------------------------------------------------
// Input projections: xg = X @ Wg + bg for g in {z,r,h}.  X:[BT,D] fp32.
// 8 waves/block share one M-tile; A strip staged once to LDS, then each wave
// computes one 16x16 output tile for all 3 gates.
// ---------------------------------------------------------------------------
__global__ void proj_kernel(const float* __restrict__ X,
                            const __bf16* __restrict__ Wzp,
                            const __bf16* __restrict__ Wrp,
                            const __bf16* __restrict__ Whp,
                            const float* __restrict__ bz,
                            const float* __restrict__ br,
                            const float* __restrict__ bh,
                            float* __restrict__ xz,
                            float* __restrict__ xr,
                            float* __restrict__ xh) {
    __shared__ alignas(32) __bf16 alds[(D_ / 32) * 32 * 16];     // 16 KB
    const int ntiles = U_ / 16;                                  // 64
    const int lane = threadIdx.x & 31;
    const int mt = blockIdx.x >> 3;                              // block-uniform
    const int nt = (blockIdx.x & 7) * 8 + (threadIdx.x >> 5);

    stage_a(alds, X, mt * 16, D_, D_ / 32);
    __syncthreads();

    v8f az = {}, ar = {}, ah = {};
#pragma unroll 4
    for (int kt = 0; kt < D_ / 32; ++kt) {
        const v16bf a = *(const v16bf*)(alds + (size_t)(kt * 32 + lane) * 16);
        const size_t tb = ((size_t)(kt * ntiles + nt) * 32 + lane) * 16;
        const v16bf bzf = *(const v16bf*)(Wzp + tb);
        const v16bf brf = *(const v16bf*)(Wrp + tb);
        const v16bf bhf = *(const v16bf*)(Whp + tb);
        if (kt + 1 < D_ / 32) {
            const size_t tn = ((size_t)((kt + 1) * ntiles + nt) * 32 + lane) * 16;
            __builtin_prefetch((const void*)(Wzp + tn), 0, 1);   // global_prefetch_b8
        }
        WMMA_BF16(az, a, bzf);
        WMMA_BF16(ar, a, brf);
        WMMA_BF16(ah, a, bhf);
    }

    // C/D layout: VGPR v, lanes 0-15 => M=v, lanes 16-31 => M=8+v; N=lane%16.
    const int col = nt * 16 + (lane & 15);
    const int rb  = mt * 16 + ((lane & 16) ? 8 : 0);
    const float vz = bz[col], vr = br[col], vh = bh[col];
#pragma unroll
    for (int v = 0; v < 8; ++v) {
        const size_t o = (size_t)(rb + v) * U_ + col;
        xz[o] = az[v] + vz;
        xr[o] = ar[v] + vr;
        xh[o] = ah[v] + vh;
    }
}

// ---------------------------------------------------------------------------
// Persistent scan kernel: all T steps in one launch.
// Grid: 32 blocks x 8 waves = 256 waves = 4 Mtiles x 64 Ntiles -> one wave per
// 16x16 tile of h; a block's 8 waves share an M-tile, so the A strip (h or
// r*h) is converted+staged to LDS once per block per phase.  z stays in
// registers across the intra-step barrier.  Two device barriers per step.
// ---------------------------------------------------------------------------
__global__ void gru_scan_kernel(float* __restrict__ hbuf,        // 2 x [B,U]
                                float* __restrict__ rbuf,        // [B,U]
                                const __bf16* __restrict__ Uzp,
                                const __bf16* __restrict__ Urp,
                                const __bf16* __restrict__ Uhp,
                                const float* __restrict__ xz,
                                const float* __restrict__ xr,
                                const float* __restrict__ xh,
                                float* __restrict__ out,
                                unsigned* __restrict__ ctr) {
    __shared__ alignas(32) __bf16 alds[(U_ / 32) * 32 * 16];     // 32 KB
    const int ntiles = U_ / 16;
    const int lane = threadIdx.x & 31;
    const int mt = blockIdx.x >> 3;                              // 0..3, uniform
    const int nt = (blockIdx.x & 7) * 8 + (threadIdx.x >> 5);    // 0..63
    const int col = nt * 16 + (lane & 15);
    const int rb  = mt * 16 + ((lane & 16) ? 8 : 0);

    unsigned bar = 0;
    for (int t = 0; t < T_; ++t) {
        const float* hp = hbuf + (size_t)(t & 1) * B_ * U_;
        float*       hn = hbuf + (size_t)((t + 1) & 1) * B_ * U_;

        // ---- phase 1: z = sigmoid(xz + h@Uz), r = sigmoid(xr + h@Ur) ----
        stage_a(alds, hp, mt * 16, U_, U_ / 32);
        __syncthreads();
        v8f az = {}, ar = {};
#pragma unroll 4
        for (int kt = 0; kt < U_ / 32; ++kt) {
            const v16bf a = *(const v16bf*)(alds + (size_t)(kt * 32 + lane) * 16);
            const size_t tb = ((size_t)(kt * ntiles + nt) * 32 + lane) * 16;
            const v16bf bz_ = *(const v16bf*)(Uzp + tb);
            const v16bf br_ = *(const v16bf*)(Urp + tb);
            WMMA_BF16(az, a, bz_);
            WMMA_BF16(ar, a, br_);
        }
        v8f zreg;
#pragma unroll
        for (int v = 0; v < 8; ++v) {
            const int b = rb + v;
            const size_t xi = ((size_t)b * T_ + t) * U_ + col;
            const size_t hi = (size_t)b * U_ + col;
            zreg[v]  = sigmoidf_(az[v] + xz[xi]);
            rbuf[hi] = sigmoidf_(ar[v] + xr[xi]);
        }
        grid_barrier(ctr, SCAN_BLOCKS * (++bar));   // r visible device-wide

        // ---- phase 2: hh = tanh(xh + (r*h)@Uh); h = z*h + (1-z)*hh ----
        stage_a_mul(alds, hp, rbuf, mt * 16, U_, U_ / 32);
        __syncthreads();
        v8f ah = {};
#pragma unroll 4
        for (int kt = 0; kt < U_ / 32; ++kt) {
            const v16bf a = *(const v16bf*)(alds + (size_t)(kt * 32 + lane) * 16);
            const size_t tb = ((size_t)(kt * ntiles + nt) * 32 + lane) * 16;
            const v16bf bh_ = *(const v16bf*)(Uhp + tb);
            WMMA_BF16(ah, a, bh_);
        }
#pragma unroll
        for (int v = 0; v < 8; ++v) {
            const int b = rb + v;
            const size_t xi = ((size_t)b * T_ + t) * U_ + col;
            const size_t hi = (size_t)b * U_ + col;
            const float hh = tanhf(ah[v] + xh[xi]);
            const float z  = zreg[v];
            const float hv = z * hp[hi] + (1.0f - z) * hh;
            hn[hi]  = hv;
            out[xi] = hv;
        }
        grid_barrier(ctr, SCAN_BLOCKS * (++bar));   // h visible device-wide
    }
}

// ---------------------------------------------------------------------------
// Host driver
// ---------------------------------------------------------------------------
extern "C" void kernel_launch(void* const* d_in, const int* in_sizes, int n_in,
                              void* d_out, int out_size, void* d_ws, size_t ws_size,
                              hipStream_t stream) {
    const float* X  = (const float*)d_in[0];
    const float* Wz = (const float*)d_in[1];
    const float* Uz = (const float*)d_in[2];
    const float* bz = (const float*)d_in[3];
    const float* Wr = (const float*)d_in[4];
    const float* Ur = (const float*)d_in[5];
    const float* br = (const float*)d_in[6];
    const float* Wh = (const float*)d_in[7];
    const float* Uh = (const float*)d_in[8];
    const float* bh = (const float*)d_in[9];
    float* out = (float*)d_out;

    // Workspace carve-up (all chunk sizes are multiples of 256 B).
    const size_t wPackElems = (size_t)(D_ / 32) * (U_ / 16) * 32 * 16;  // bf16
    const size_t uPackElems = (size_t)(U_ / 32) * (U_ / 16) * 32 * 16;  // bf16
    char* p = (char*)d_ws;
    __bf16* Wzp = (__bf16*)p; p += wPackElems * sizeof(__bf16);
    __bf16* Wrp = (__bf16*)p; p += wPackElems * sizeof(__bf16);
    __bf16* Whp = (__bf16*)p; p += wPackElems * sizeof(__bf16);
    __bf16* Uzp = (__bf16*)p; p += uPackElems * sizeof(__bf16);
    __bf16* Urp = (__bf16*)p; p += uPackElems * sizeof(__bf16);
    __bf16* Uhp = (__bf16*)p; p += uPackElems * sizeof(__bf16);
    float* hbuf = (float*)p;  p += (size_t)2 * B_ * U_ * sizeof(float);
    float* rbuf = (float*)p;  p += (size_t)B_ * U_ * sizeof(float);
    float* xz   = (float*)p;  p += (size_t)BT_ * U_ * sizeof(float);
    float* xr   = (float*)p;  p += (size_t)BT_ * U_ * sizeof(float);
    float* xh   = (float*)p;  p += (size_t)BT_ * U_ * sizeof(float);
    unsigned* ctr = (unsigned*)p;

    // h0 = 0 and barrier counter = 0 (re-done every call -> graph-replay safe).
    hipMemsetAsync(hbuf, 0, (size_t)2 * B_ * U_ * sizeof(float), stream);
    hipMemsetAsync(ctr, 0, 256, stream);

    // Pack weights to bf16 WMMA-B fragment layout.
    {
        const int wThreads = (D_ / 32) * (U_ / 16) * 32;   // 32768
        const int uThreads = (U_ / 32) * (U_ / 16) * 32;   // 65536
        pack_b_kernel<<<wThreads / 256, 256, 0, stream>>>(Wz, Wzp, D_, U_);
        pack_b_kernel<<<wThreads / 256, 256, 0, stream>>>(Wr, Wrp, D_, U_);
        pack_b_kernel<<<wThreads / 256, 256, 0, stream>>>(Wh, Whp, D_, U_);
        pack_b_kernel<<<uThreads / 256, 256, 0, stream>>>(Uz, Uzp, U_, U_);
        pack_b_kernel<<<uThreads / 256, 256, 0, stream>>>(Ur, Urp, U_, U_);
        pack_b_kernel<<<uThreads / 256, 256, 0, stream>>>(Uh, Uhp, U_, U_);
    }

    // Input projections: one block per (Mtile, 8-Ntile group).
    {
        const int blocks = (BT_ / 16) * 8;                 // 16384
        proj_kernel<<<blocks, BLOCK_THREADS, 0, stream>>>(X, Wzp, Wrp, Whp,
                                                          bz, br, bh, xz, xr, xh);
    }

    // Entire 512-step scan in one persistent launch (32 co-resident blocks).
    gru_scan_kernel<<<SCAN_BLOCKS, BLOCK_THREADS, 0, stream>>>(hbuf, rbuf,
                                                               Uzp, Urp, Uhp,
                                                               xz, xr, xh, out, ctr);
}
